// DynamicConv2DMobius_38774964748896
// MI455X (gfx1250) — matrix operand order repaired and checked by reading, not compiled, not run on hardware
//
#include <hip/hip_runtime.h>
#include <hip/hip_bf16.h>

typedef __attribute__((ext_vector_type(16))) _Float16 v16h;
typedef __attribute__((ext_vector_type(8)))  _Float16 v8h;
typedef __attribute__((ext_vector_type(8)))  float    v8f;

#define OC 256
#define IC 128
#define HD 768
#define WD 384
#define CTX 256
#define NB 32
#define KTOT 1152   // IC*9

// ---------- workspace layout (bytes) ----------
#define OFF_WRE  ((size_t)0)           // 32*16*16 f32
#define OFF_WIM  ((size_t)32768)
#define OFF_SC   ((size_t)65536)       // s_coef [32,16,16] f32
#define OFF_U16  ((size_t)98304)       // U_cat f16 [768,256]
#define OFF_SVT  ((size_t)491520)      // sVT f16 [32,384,256]
#define OFF_WA   ((size_t)6782976)     // wA f16 [32,256,1152]
#define OFF_XP   ((size_t)25657344)    // xp16 f16 NHWC [32,66,66,128]

__device__ __forceinline__ v16h mk16(v8h lo, v8h hi) {
    v16h r;
#pragma unroll
    for (int t = 0; t < 8; ++t) { r[t] = lo[t]; r[t + 8] = hi[t]; }
    return r;
}

// ---- P1: Mobius coefficients: (b,s,r) -> w_re, w_im --------------------
__global__ void k_mobius(const float* __restrict__ ctx, const float* __restrict__ Wm,
                         const float* __restrict__ z, float* __restrict__ wre,
                         float* __restrict__ wim) {
    int idx = blockIdx.x * blockDim.x + threadIdx.x;       // 8192 = 32*16*16
    int r = idx & 15, s = (idx >> 4) & 15, b = idx >> 8;
    float m[8] = {0.f,0.f,0.f,0.f,0.f,0.f,0.f,0.f};
    const float* cb = ctx + b * CTX;
    for (int c = 0; c < CTX; ++c) {
        float cv = cb[c];
        const float* wp = Wm + (((size_t)c * 16 + s) * 16 + r) * 8;
#pragma unroll
        for (int j = 0; j < 8; ++j) m[j] = fmaf(cv, wp[j], m[j]);
    }
    float zre = z[(s * 16 + r) * 2 + 0], zim = z[(s * 16 + r) * 2 + 1];
    float nre = m[0] * zre - m[1] * zim + m[2];
    float nim = m[0] * zim + m[1] * zre + m[3];
    float dre = m[4] * zre - m[5] * zim + m[6];
    float dim = m[4] * zim + m[5] * zre + m[7];
    float den2 = dre * dre + dim * dim + 1e-3f;
    wre[idx] = (nre * dre + nim * dim) / den2;
    wim[idx] = (nim * dre - nre * dim) / den2;
}

// ---- P2: s_coef[b,s,q] = sum_r wre*A_re + wim*A_im ---------------------
__global__ void k_scoef(const float* __restrict__ wre, const float* __restrict__ wim,
                        const float* __restrict__ Are, const float* __restrict__ Aim,
                        float* __restrict__ sc) {
    int idx = blockIdx.x * blockDim.x + threadIdx.x;       // 8192 = (b,s,q)
    int q = idx & 15, s = (idx >> 4) & 15, b = idx >> 8;
    float acc = 0.f;
#pragma unroll
    for (int r = 0; r < 16; ++r) {
        int bi = (b * 16 + s) * 16 + r;
        int ai = (s * 16 + r) * 16 + q;
        acc = fmaf(wre[bi], Are[ai], acc);
        acc = fmaf(wim[bi], Aim[ai], acc);
    }
    sc[idx] = acc;
}

// ---- P3: U_cat16[h][kk] = f16(U[s,h,q]),  kk = s*16+q ------------------
__global__ void k_u16(const float* __restrict__ U, _Float16* __restrict__ U16) {
    int idx = blockIdx.x * blockDim.x + threadIdx.x;       // 768*256
    int kk = idx & 255, h = idx >> 8;
    int s = kk >> 4, q = kk & 15;
    U16[idx] = (_Float16)U[((size_t)(s * HD + h)) * 16 + q];
}

// ---- P4: sVT16[b][w][kk] = f16(s_coef[b,s,q] * V[s,q,w]) ---------------
__global__ void k_svt(const float* __restrict__ sc, const float* __restrict__ V,
                      _Float16* __restrict__ svt) {
    int idx = blockIdx.x * blockDim.x + threadIdx.x;       // 32*384*256
    int kk = idx & 255;
    int w = (idx >> 8) % WD;
    int b = idx / (WD * 256);
    int s = kk >> 4, q = kk & 15;
    svt[idx] = (_Float16)(sc[(b * 16 + s) * 16 + q] * V[(size_t)(s * 16 + q) * WD + w]);
}

// ---- P5: padded input, NHWC f16: xp16[b][yy][xx][i] --------------------
__global__ void k_xpad(const float* __restrict__ x, const float* __restrict__ padv,
                       const float* __restrict__ offv, _Float16* __restrict__ xp) {
    int idx = blockIdx.x * blockDim.x + threadIdx.x;       // 32*128*66*66 (xx fastest)
    int xx = idx % 66;
    int yy = (idx / 66) % 66;
    int i  = (idx / 4356) % IC;
    int b  = idx / 557568;
    float v;
    if (yy >= 1 && yy <= 64 && xx >= 1 && xx <= 64)
        v = x[(((size_t)b * IC + i) * 64 + (yy - 1)) * 64 + (xx - 1)] + offv[0];
    else
        v = padv[0];
    xp[(((size_t)b * 66 + yy) * 66 + xx) * IC + i] = (_Float16)v;
}

// ---- GEMM2 + fused gather: wA[b][o][k] -------------------------------
// dw[b] = U_cat[768x256] @ sVT[b]^T[256x384] + bias_dw, scattered into conv
// weight layout k = (kh*3+kw)*128 + i.
__global__ void k_dwgemm(const _Float16* __restrict__ U16, const _Float16* __restrict__ svt,
                         const float* __restrict__ bias_dw, _Float16* __restrict__ wA) {
    int gtid = blockIdx.x * blockDim.x + threadIdx.x;
    int wave = gtid >> 5;                                  // 36864 waves total
    int lane = threadIdx.x & 31;
    int b = wave / 1152;
    int rem = wave % 1152;
    int mt = rem / 24, nt = rem % 24;
    int n = lane & 15;
    int off8  = (lane & 16) ? 8 : 0;
    int off16 = (lane & 16) ? 16 : 0;

    const _Float16* Ap = U16 + (size_t)(mt * 16 + n) * 256;
    const _Float16* Bp = svt + ((size_t)b * WD + nt * 16 + n) * 256 + off16;

    v8f acc = (v8f)0.0f;
#pragma unroll
    for (int kc = 0; kc < 8; ++kc) {
        int kb = kc * 32;
        v16h a = *(const v16h*)(Ap + kb + off16);
        v16h bf = *(const v16h*)(Bp + kb);
        acc = __builtin_amdgcn_wmma_f32_16x16x32_f16(false, a, false, bf,
                                                     (short)0, acc, false, false);
    }
#pragma unroll
    for (int v = 0; v < 8; ++v) {
        int hh = mt * 16 + v + off8;
        int ww = nt * 16 + n;
        float val = acc[v] + bias_dw[hh * WD + ww];
        int bi = hh / 3, kh = hh - 3 * bi;
        int bj = ww / 3, kw = ww - 3 * bj;
        int kblk = bi * 128 + bj;                          // = i*256 + o
        int o = kblk & 255, i = kblk >> 8;
        int k = (kh * 3 + kw) * 128 + i;
        wA[((size_t)b * OC + o) * KTOT + k] = (_Float16)val;
    }
}

// ---- Conv as implicit GEMM with WMMA ----------------------------------
// Block: 256 threads = 8 waves; grid = 32*64 (b, y). Wave w handles OC tiles
// {2w, 2w+1} x 4 pixel tiles (full row of 64 x). K = 1152 in 36 chunks of 32.
__global__ void k_conv(const _Float16* __restrict__ xp, const _Float16* __restrict__ wA,
                       const float* __restrict__ bias_c, float* __restrict__ out) {
    int b = blockIdx.x >> 6;
    int y = blockIdx.x & 63;
    int wave = threadIdx.x >> 5;
    int lane = threadIdx.x & 31;
    int n = lane & 15;
    int off8  = (lane & 16) ? 8 : 0;
    int off16 = (lane & 16) ? 16 : 0;

    const _Float16* Arow0 = wA + ((size_t)b * OC + wave * 32 + n) * KTOT;
    const _Float16* Arow1 = Arow0 + (size_t)16 * KTOT;
    const _Float16* xpb = xp + (size_t)b * 66 * 66 * IC;

    v8f acc[2][4];
#pragma unroll
    for (int ot = 0; ot < 2; ++ot)
#pragma unroll
        for (int t4 = 0; t4 < 4; ++t4) acc[ot][t4] = (v8f)0.0f;

    for (int c = 0; c < 36; ++c) {
        int kb = c * 32;
        int pos = c >> 2;                 // kh*3+kw
        int ibase = (c & 3) * 32;
        int kh = pos / 3, kw = pos - 3 * kh;

        v16h a0 = *(const v16h*)(Arow0 + kb + off16);
        v16h a1 = *(const v16h*)(Arow1 + kb + off16);

        const _Float16* xrow = xpb + (size_t)(y + kh) * 66 * IC;
#pragma unroll
        for (int t4 = 0; t4 < 4; ++t4) {
            int xcol = t4 * 16 + n + kw;
            v16h bf = *(const v16h*)(xrow + (size_t)xcol * IC + ibase + off16);
            acc[0][t4] = __builtin_amdgcn_wmma_f32_16x16x32_f16(false, a0, false, bf,
                                                                (short)0, acc[0][t4], false, false);
            acc[1][t4] = __builtin_amdgcn_wmma_f32_16x16x32_f16(false, a1, false, bf,
                                                                (short)0, acc[1][t4], false, false);
        }
    }

#pragma unroll
    for (int ot = 0; ot < 2; ++ot) {
#pragma unroll
        for (int v = 0; v < 8; ++v) {
            int oc = wave * 32 + ot * 16 + v + off8;
            float bc = bias_c[oc];
            float* orow = out + (((size_t)b * OC + oc) * 64 + y) * 64;
#pragma unroll
            for (int t4 = 0; t4 < 4; ++t4)
                orow[t4 * 16 + n] = acc[ot][t4][v] + bc;
        }
    }
}

extern "C" void kernel_launch(void* const* d_in, const int* in_sizes, int n_in,
                              void* d_out, int out_size, void* d_ws, size_t ws_size,
                              hipStream_t stream) {
    const float* x       = (const float*)d_in[0];
    const float* context = (const float*)d_in[1];
    const float* W_mob   = (const float*)d_in[2];
    const float* z       = (const float*)d_in[3];
    const float* A_re    = (const float*)d_in[4];
    const float* A_im    = (const float*)d_in[5];
    const float* U       = (const float*)d_in[6];
    const float* V       = (const float*)d_in[7];
    const float* bias_dw = (const float*)d_in[8];
    const float* bias_cv = (const float*)d_in[9];
    const float* pad_val = (const float*)d_in[10];
    const float* off_val = (const float*)d_in[11];
    // d_in[12] (dw_index) is unused: the permutation is computed analytically.

    char* ws = (char*)d_ws;
    float*    wre  = (float*)(ws + OFF_WRE);
    float*    wim  = (float*)(ws + OFF_WIM);
    float*    sc   = (float*)(ws + OFF_SC);
    _Float16* U16  = (_Float16*)(ws + OFF_U16);
    _Float16* svt  = (_Float16*)(ws + OFF_SVT);
    _Float16* wA   = (_Float16*)(ws + OFF_WA);
    _Float16* xp16 = (_Float16*)(ws + OFF_XP);
    float*    out  = (float*)d_out;

    k_mobius<<<32, 256, 0, stream>>>(context, W_mob, z, wre, wim);
    k_scoef <<<32, 256, 0, stream>>>(wre, wim, A_re, A_im, sc);
    k_u16   <<<768, 256, 0, stream>>>(U, U16);
    k_svt   <<<12288, 256, 0, stream>>>(sc, V, svt);
    k_xpad  <<<69696, 256, 0, stream>>>(x, pad_val, off_val, xp16);
    k_dwgemm<<<4608, 256, 0, stream>>>(U16, svt, bias_dw, wA);
    k_conv  <<<2048, 256, 0, stream>>>(xp16, wA, bias_cv, out);
}
